// GNNActor_62130996904181
// MI455X (gfx1250) — compile-verified, matrix operand order checked by reading, not run on hardware
//
#include <hip/hip_runtime.h>
#include <hip/hip_bf16.h>

typedef __attribute__((ext_vector_type(16))) _Float16 v16h;
typedef __attribute__((ext_vector_type(8)))  float    v8f;

#define N_NODES 20000
#define N_EDGES 320000

// ---------------------------------------------------------------------------
// Weight repack: fp32 row-major W[K,N] -> f16 WMMA B-fragment order.
// P[((ct*(K/32)+kt)*32 + lane)*16 + e], col = ct*16+(lane&15),
// k = kt*32 + (lane>>4)*8 + (e<8 ? e : 16+(e-8)).
// ---------------------------------------------------------------------------
__global__ void pack_w_f16(const float* __restrict__ W, _Float16* __restrict__ P,
                           int K, int N)
{
    int gid = blockIdx.x * blockDim.x + threadIdx.x;
    int K32 = K >> 5;
    int total = (N >> 4) * K32 * 32;
    if (gid >= total) return;
    int lane = gid & 31;
    int tile = gid >> 5;
    int kt = tile % K32;
    int ct = tile / K32;
    int col = ct * 16 + (lane & 15);
    int kb  = kt * 32 + (lane >> 4) * 8;
    _Float16* __restrict__ p = P + (size_t)gid * 16;
#pragma unroll
    for (int e = 0; e < 8; ++e) p[e]     = (_Float16)W[(size_t)(kb + e) * N + col];
#pragma unroll
    for (int e = 0; e < 8; ++e) p[8 + e] = (_Float16)W[(size_t)(kb + 16 + e) * N + col];
}

// ---------------------------------------------------------------------------
// WMMA GEMM: C[M,N] = A[M,K] @ W[K,N] (+bias,+relu). A fp32 row-major, W
// pre-packed f16 fragments. Block = 256 thr = 8 waves; each wave: 16 rows x
// NT col-tiles of 16. All NT B fragments issued before the WMMA group so the
// clause pipelines; no per-tile branches (NT is compile-time).
// ---------------------------------------------------------------------------
template <int NT, bool BIAS, bool RELU>
__global__ __launch_bounds__(256)
void gemm_wmma_f16(const float* __restrict__ A, const _Float16* __restrict__ Bp,
                   const float* __restrict__ bias, float* __restrict__ C,
                   int M, int K, int N)
{
    const int lane = threadIdx.x & 31;
    const int wave = threadIdx.x >> 5;
    const int half = lane >> 4;
    const int l16  = lane & 15;
    const int K32  = K >> 5;

    const int rowBase = blockIdx.y * 128 + wave * 16;     // wave-uniform
    const int colBase = blockIdx.x * (NT * 16);
    const int ct0     = colBase >> 4;
    if (rowBase >= M) return;

    int arow = rowBase + l16;
    if (arow >= M) arow = M - 1;                          // clamp; store guarded
    const float* __restrict__ Arow = A + (size_t)arow * K;
    const v16h* __restrict__ Bf = (const v16h*)Bp;

    v8f acc[NT];
#pragma unroll
    for (int t = 0; t < NT; ++t) acc[t] = v8f{0};

    for (int kt = 0; kt < K32; ++kt) {
        const int ka = kt * 32 + half * 8;

        // All NT B fragments first: 2*NT global_load_b128 in one clause
        v16h bf[NT];
#pragma unroll
        for (int t = 0; t < NT; ++t)
            bf[t] = Bf[((size_t)(ct0 + t) * K32 + kt) * 32 + lane];

        // A fragment: two contiguous 8-float runs -> 4x global_load_b128
        float4 a0 = *(const float4*)(Arow + ka);
        float4 a1 = *(const float4*)(Arow + ka + 4);
        float4 a2 = *(const float4*)(Arow + ka + 16);
        float4 a3 = *(const float4*)(Arow + ka + 20);
        if (kt + 1 < K32)
            __builtin_prefetch(Arow + ka + 32, 0, 3);     // global_prefetch_b8

        v16h afrag;
        afrag[0]  = (_Float16)a0.x; afrag[1]  = (_Float16)a0.y;
        afrag[2]  = (_Float16)a0.z; afrag[3]  = (_Float16)a0.w;
        afrag[4]  = (_Float16)a1.x; afrag[5]  = (_Float16)a1.y;
        afrag[6]  = (_Float16)a1.z; afrag[7]  = (_Float16)a1.w;
        afrag[8]  = (_Float16)a2.x; afrag[9]  = (_Float16)a2.y;
        afrag[10] = (_Float16)a2.z; afrag[11] = (_Float16)a2.w;
        afrag[12] = (_Float16)a3.x; afrag[13] = (_Float16)a3.y;
        afrag[14] = (_Float16)a3.z; afrag[15] = (_Float16)a3.w;

#pragma unroll
        for (int t = 0; t < NT; ++t)
            acc[t] = __builtin_amdgcn_wmma_f32_16x16x32_f16(
                false, afrag, false, bf[t], (short)0, acc[t], false, false);
    }

    // C store: lane(0-15): N=l16; VGPR v -> M = half*8 + v
#pragma unroll
    for (int t = 0; t < NT; ++t) {
        int col = colBase + t * 16 + l16;
#pragma unroll
        for (int v = 0; v < 8; ++v) {
            int row = rowBase + half * 8 + v;
            if (row >= M) continue;
            float val = acc[t][v];
            if (BIAS) val += bias[col];
            if (RELU) val = fmaxf(val, 0.0f);
            C[(size_t)row * N + col] = val;
        }
    }
}

// ---------------------------------------------------------------------------
// Elementwise / graph kernels
// ---------------------------------------------------------------------------
__global__ void fill_f32v4(float4* __restrict__ p, float v, long long n4)
{
    long long i = (long long)blockIdx.x * blockDim.x + threadIdx.x;
    if (i < n4) p[i] = make_float4(v, v, v, v);
}

__global__ void fill_f32(float* __restrict__ p, float v, long long n)
{
    long long i = (long long)blockIdx.x * blockDim.x + threadIdx.x;
    if (i < n) p[i] = v;
}

__global__ void deg_count(const long long* __restrict__ dst, float* __restrict__ deg, int E)
{
    int i = blockIdx.x * blockDim.x + threadIdx.x;
    if (i < E) atomicAdd(&deg[(int)dst[i]], 1.0f);
}

__global__ void rsqrt_inplace(float* __restrict__ d, int n)
{
    int i = blockIdx.x * blockDim.x + threadIdx.x;
    if (i < n) d[i] = rsqrtf(d[i]);
}

// One wave per edge; each lane reads float4 of h[src] (global_load_b128) and
// issues 4 global_atomic_add_f32 into agg[dst]. F % 128 == 0 (512/256/128).
__global__ void scatter_edges(const float* __restrict__ h,
                              const long long* __restrict__ src,
                              const long long* __restrict__ dst,
                              const float* __restrict__ dis,
                              float* __restrict__ agg, int E, int F)
{
    int gid  = blockIdx.x * blockDim.x + threadIdx.x;
    int e    = gid >> 5;
    int lane = gid & 31;
    if (e >= E) return;
    int s = (int)src[e];
    int d = (int)dst[e];
    float norm = dis[s] * dis[d];
    const float4* __restrict__ hs = (const float4*)(h + (size_t)s * F);
    float*        __restrict__ ad = agg + (size_t)d * F;
    const int F4 = F >> 2;
    for (int c = lane; c < F4; c += 32) {
        float4 v = hs[c];
        int f = c * 4;
        atomicAdd(&ad[f + 0], v.x * norm);
        atomicAdd(&ad[f + 1], v.y * norm);
        atomicAdd(&ad[f + 2], v.z * norm);
        atomicAdd(&ad[f + 3], v.w * norm);
    }
}

// out = relu(agg + h*dis^2 + bias) [+ res], float4 per thread (out may alias agg)
__global__ void finalize_gcn(const float4* __restrict__ agg, const float4* __restrict__ h,
                             const float* __restrict__ dis, const float4* __restrict__ bias,
                             const float4* __restrict__ res, float4* __restrict__ out,
                             int Nn, int F, int doRes)
{
    long long idx = (long long)blockIdx.x * blockDim.x + threadIdx.x;
    const int F4 = F >> 2;
    long long total = (long long)Nn * F4;
    if (idx >= total) return;
    int i = (int)(idx / F4);
    int r = (int)(idx % F4);
    float di = dis[i];
    float d2 = di * di;
    float4 a = agg[idx], hh = h[idx], bb = bias[r];
    float4 v;
    v.x = fmaxf(a.x + hh.x * d2 + bb.x, 0.0f);
    v.y = fmaxf(a.y + hh.y * d2 + bb.y, 0.0f);
    v.z = fmaxf(a.z + hh.z * d2 + bb.z, 0.0f);
    v.w = fmaxf(a.w + hh.w * d2 + bb.w, 0.0f);
    if (doRes) {
        float4 rr = res[idx];
        v.x += rr.x; v.y += rr.y; v.z += rr.z; v.w += rr.w;
    }
    out[idx] = v;
}

// Final 32 -> 2 head; d_out = (h[:,0] | h[:,1]) concatenated.
__global__ void head2(const float* __restrict__ m, const float* __restrict__ W,
                      const float* __restrict__ b, float* __restrict__ out, int Nn)
{
    int i = blockIdx.x * blockDim.x + threadIdx.x;
    if (i >= Nn) return;
    const float* __restrict__ mi = m + (size_t)i * 32;
    float a0 = b[0], a1 = b[1];
#pragma unroll
    for (int k = 0; k < 32; ++k) {
        float v = mi[k];
        a0 += v * W[k * 2 + 0];
        a1 += v * W[k * 2 + 1];
    }
    out[i]      = a0;
    out[Nn + i] = a1;
}

// ---------------------------------------------------------------------------
// Launch
// ---------------------------------------------------------------------------
extern "C" void kernel_launch(void* const* d_in, const int* in_sizes, int n_in,
                              void* d_out, int out_size, void* d_ws, size_t ws_size,
                              hipStream_t stream)
{
    (void)in_sizes; (void)n_in; (void)out_size; (void)ws_size;

    const float*     x  = (const float*)d_in[0];
    const long long* ei = (const long long*)d_in[1];
    const float* W1 = (const float*)d_in[2];  const float* b1 = (const float*)d_in[3];
    const float* W2 = (const float*)d_in[4];  const float* b2 = (const float*)d_in[5];
    const float* W3 = (const float*)d_in[6];  const float* b3 = (const float*)d_in[7];
    const float* lw1 = (const float*)d_in[8];  const float* lb1 = (const float*)d_in[9];
    const float* lw2 = (const float*)d_in[10]; const float* lb2 = (const float*)d_in[11];
    const float* lw3 = (const float*)d_in[12]; const float* lb3 = (const float*)d_in[13];
    const float* lw4 = (const float*)d_in[14]; const float* lb4 = (const float*)d_in[15];

    float* out = (float*)d_out;
    float* ws  = (float*)d_ws;

    const int Nn = N_NODES, E = N_EDGES;
    const long long* src = ei;        // edge_index[0]
    const long long* dst = ei + E;    // edge_index[1]

    float* dis  = ws;                             // [20000] (padded to 20480)
    float* buf0 = ws + 20480;                     // [20000*512]
    float* buf1 = buf0 + (size_t)Nn * 512;        // [20000*512]
    _Float16* pk = (_Float16*)(buf1 + (size_t)Nn * 512);  // packed weights

    _Float16* pW1  = pk;                    // 128*512 = 65536
    _Float16* pW2  = pW1 + 65536;           // 512*256 = 131072
    _Float16* pW3  = pW2 + 131072;          // 256*128 = 32768
    _Float16* pLW1 = pW3 + 32768;           // 128*128 = 16384
    _Float16* pLW2 = pLW1 + 16384;          // 128*64  = 8192
    _Float16* pLW3 = pLW2 + 8192;           // 64*32   = 2048

    const int T = 256;
    auto blocks = [](long long n, int t) { return (unsigned)((n + t - 1) / t); };

    // --- repack all weights to f16 fragment order (tiny; L2-resident)
    auto pack = [&](const float* W, _Float16* P, int K, int N) {
        long long tot = (long long)(N >> 4) * (K >> 5) * 32;
        pack_w_f16<<<blocks(tot, T), T, 0, stream>>>(W, P, K, N);
    };
    pack(W1, pW1, 128, 512);
    pack(W2, pW2, 512, 256);
    pack(W3, pW3, 256, 128);
    pack(lw1, pLW1, 128, 128);
    pack(lw2, pLW2, 128, 64);
    pack(lw3, pLW3, 64, 32);

    // --- degrees (start at 1.0 for self loop), dis = rsqrt(deg)
    fill_f32<<<blocks(Nn, T), T, 0, stream>>>(dis, 1.0f, Nn);
    deg_count<<<blocks(E, T), T, 0, stream>>>(dst, dis, E);
    rsqrt_inplace<<<blocks(Nn, T), T, 0, stream>>>(dis, Nn);

    // --- one GCN layer (Fout in {512,256,128} -> NT=4)
    auto gcn = [&](const float* in, int Fin, int Fout, const _Float16* Wp, const float* b,
                   const float* res, float* hbuf, float* aggbuf) {
        dim3 g((Fout + 63) / 64, (Nn + 127) / 128);
        gemm_wmma_f16<4, false, false><<<g, 256, 0, stream>>>(in, Wp, nullptr, hbuf,
                                                              Nn, Fin, Fout);
        long long tot4 = (long long)Nn * (Fout >> 2);
        fill_f32v4<<<blocks(tot4, T), T, 0, stream>>>((float4*)aggbuf, 0.0f, tot4);
        scatter_edges<<<blocks((long long)E * 32, T), T, 0, stream>>>(hbuf, src, dst, dis,
                                                                      aggbuf, E, Fout);
        finalize_gcn<<<blocks(tot4, T), T, 0, stream>>>((const float4*)aggbuf, (const float4*)hbuf,
                                                        dis, (const float4*)b, (const float4*)res,
                                                        (float4*)aggbuf, Nn, Fout, res != nullptr);
    };

    gcn(x,    128, 512, pW1, b1, nullptr, buf0, buf1);  // out1 in buf1 [N,512]
    gcn(buf1, 512, 256, pW2, b2, nullptr, buf0, buf1);  // out2 in buf1 [N,256]
    gcn(buf1, 256, 128, pW3, b3, x,       buf0, buf1);  // out3+x in buf1 [N,128]

    // --- MLP head (bias+relu fused into WMMA epilogue)
    {
        dim3 g1(2, (Nn + 127) / 128);   // N=128 -> 2 blocks of 64 cols
        gemm_wmma_f16<4, true, true><<<g1, 256, 0, stream>>>(buf1, pLW1, lb1, buf0, Nn, 128, 128);
        dim3 g2(1, (Nn + 127) / 128);   // N=64 -> 4 tiles
        gemm_wmma_f16<4, true, true><<<g2, 256, 0, stream>>>(buf0, pLW2, lb2, buf1, Nn, 128, 64);
        dim3 g3(1, (Nn + 127) / 128);   // N=32 -> 2 tiles
        gemm_wmma_f16<2, true, true><<<g3, 256, 0, stream>>>(buf1, pLW3, lb3, buf0, Nn, 64, 32);
        head2<<<blocks(Nn, T), T, 0, stream>>>(buf0, lw4, lb4, out, Nn);
    }
}
